// Attention_90958817395388
// MI455X (gfx1250) — matrix operand order repaired and checked by reading, not compiled, not run on hardware
//
#include <hip/hip_runtime.h>

// Problem constants (match reference setup_inputs)
#define NB   2
#define SEQ  2048
#define DIM  2048
#define NH   16
#define HD   128
#define BS   (NB*SEQ)

typedef __attribute__((ext_vector_type(16))) __bf16        v16bf;
typedef __attribute__((ext_vector_type(8)))  float         v8f;
typedef __attribute__((ext_vector_type(4)))  float         v4f;
typedef __attribute__((ext_vector_type(4)))  unsigned int  v4u;

union FragU { struct { v4u lo, hi; } u; v16bf v; };

static __device__ __forceinline__ unsigned short f2bf(float f) {
  unsigned int u = __float_as_uint(f);
  u += 0x7FFFu + ((u >> 16) & 1u);          // round-to-nearest-even
  return (unsigned short)(u >> 16);
}
static __device__ __forceinline__ float bf2f(unsigned short h) {
  return __uint_as_float(((unsigned int)h) << 16);
}

// 512B/wave async copy global -> LDS (CDNA5, tracked by ASYNCcnt).
static __device__ __forceinline__ void async_cp16(unsigned lds_off, const void* g) {
  unsigned long long ga = (unsigned long long)g;
  asm volatile("global_load_async_to_lds_b128 %0, %1, off"
               :: "v"(lds_off), "v"(ga) : "memory");
}
static __device__ __forceinline__ void wait_async0() {
  asm volatile("s_wait_asynccnt 0x0" ::: "memory");
}
static __device__ __forceinline__ unsigned lds_off(const void* p) {
  return (unsigned)(unsigned long long)p;   // low 32b of generic LDS addr = offset
}

static __device__ __forceinline__ v16bf load_frag(const unsigned short* p,
                                                  int off_lo, int off_hi) {
  FragU f;
  f.u.lo = *(const v4u*)(p + off_lo);
  f.u.hi = *(const v4u*)(p + off_hi);
  return f.v;
}

static __device__ __forceinline__ v8f wmma_bf16(v16bf a, v16bf b, v8f c) {
  return __builtin_amdgcn_wmma_f32_16x16x32_bf16(
      false, a, false, b, (short)0, c, false, false);
}

// ---------------------------------------------------------------------------
// fp32 -> bf16 bulk conversion (one pass; removes conversions from hot loops)
// ---------------------------------------------------------------------------
__global__ void __launch_bounds__(256)
convert_bf16(const float* __restrict__ src, unsigned short* __restrict__ dst, int n8) {
  int i = blockIdx.x * blockDim.x + threadIdx.x;
  if (i >= n8) return;
  v4f a = *(const v4f*)(src + (size_t)i * 8);
  v4f b = *(const v4f*)(src + (size_t)i * 8 + 4);
  v4u o;
  o[0] = (unsigned)f2bf(a[0]) | ((unsigned)f2bf(a[1]) << 16);
  o[1] = (unsigned)f2bf(a[2]) | ((unsigned)f2bf(a[3]) << 16);
  o[2] = (unsigned)f2bf(b[0]) | ((unsigned)f2bf(b[1]) << 16);
  o[3] = (unsigned)f2bf(b[2]) | ((unsigned)f2bf(b[3]) << 16);
  *(v4u*)(dst + (size_t)i * 8) = o;
}

// ---------------------------------------------------------------------------
// GEMM: C[M,N] = A[M,K] * W[N,K]^T,  A/W bf16, C fp32 or bf16.
// 128x128 tile, K-step 64, 8 waves x (64x32).  Staging via async-to-LDS.
// ---------------------------------------------------------------------------
#define GT_M 128
#define GT_N 128
#define GT_K 64
#define GLDA 72   // LDS row stride halves (64 + 8 pad; 144B = 9*16B)

template <bool OUT_BF16>
__global__ void __launch_bounds__(256)
gemm_bf16(const unsigned short* __restrict__ A, const unsigned short* __restrict__ W,
          void* __restrict__ Cv, int M, int N, int K) {
  __shared__ alignas(16) unsigned short As[GT_M * GLDA];
  __shared__ alignas(16) unsigned short Ws[GT_N * GLDA];
  (void)M;

  const int tid  = threadIdx.x;
  const int lane = tid & 31;
  const int w    = tid >> 5;
  const int wm   = w & 1;
  const int wn   = w >> 1;
  const int lr   = lane & 15;
  const int lh   = lane >> 4;
  const int nbase = blockIdx.x * GT_N;
  const int mbase = blockIdx.y * GT_M;

  v8f zero = {};
  v8f acc[4][2];
#pragma unroll
  for (int mi = 0; mi < 4; ++mi)
#pragma unroll
    for (int ni = 0; ni < 2; ++ni) acc[mi][ni] = zero;

  for (int kb = 0; kb < K; kb += GT_K) {
    __syncthreads();   // previous tile fully consumed
    // Stage: 128 rows x 64 halves (8 x 16B chunks per row) per tile.
#pragma unroll
    for (int t = 0; t < 4; ++t) {
      int idx = t * 256 + tid;          // 0..1023
      int r   = idx >> 3;
      int c8  = idx & 7;
      async_cp16(lds_off(As + r * GLDA + c8 * 8),
                 A + (size_t)(mbase + r) * K + kb + c8 * 8);
      async_cp16(lds_off(Ws + r * GLDA + c8 * 8),
                 W + (size_t)(nbase + r) * K + kb + c8 * 8);
    }
    if (kb + GT_K < K) {
      __builtin_prefetch(A + (size_t)(mbase + (tid >> 1)) * K + kb + GT_K, 0, 1);
      __builtin_prefetch(W + (size_t)(nbase + (tid >> 1)) * K + kb + GT_K, 0, 1);
    }
    wait_async0();
    __syncthreads();

#pragma unroll
    for (int ks = 0; ks < 2; ++ks) {
      const unsigned short* abase = As + (wm * 64) * GLDA + ks * 32;
      v16bf af[4];
#pragma unroll
      for (int mi = 0; mi < 4; ++mi) {
        const unsigned short* p = abase + (mi * 16 + lr) * GLDA;
        af[mi] = load_frag(p, lh * 8, 16 + lh * 8);
      }
#pragma unroll
      for (int ni = 0; ni < 2; ++ni) {
        const unsigned short* p =
            Ws + (wn * 32 + ni * 16 + lr) * GLDA + ks * 32;
        v16bf bf = load_frag(p, lh * 8, 16 + lh * 8);
#pragma unroll
        for (int mi = 0; mi < 4; ++mi)
          acc[mi][ni] = wmma_bf16(af[mi], bf, acc[mi][ni]);
      }
    }
  }

#pragma unroll
  for (int mi = 0; mi < 4; ++mi)
#pragma unroll
    for (int ni = 0; ni < 2; ++ni) {
      int col  = nbase + wn * 32 + ni * 16 + lr;
      int rowb = mbase + wm * 64 + mi * 16 + lh * 8;
#pragma unroll
      for (int r = 0; r < 8; ++r) {
        if constexpr (OUT_BF16) {
          ((unsigned short*)Cv)[(size_t)(rowb + r) * N + col] = f2bf(acc[mi][ni][r]);
        } else {
          ((float*)Cv)[(size_t)(rowb + r) * N + col] = acc[mi][ni][r];
        }
      }
    }
}

// ---------------------------------------------------------------------------
// RoPE on bf16 Q and K in place.  One thread per (b,s,h,pair).
// ---------------------------------------------------------------------------
__global__ void __launch_bounds__(256)
rope_qk(unsigned short* __restrict__ Q, unsigned short* __restrict__ Kt,
        const float* __restrict__ fc, const float* __restrict__ fs) {
  const int HALF = HD / 2;
  int pid = blockIdx.x * blockDim.x + threadIdx.x;
  int i  = pid & (HALF - 1);
  int t  = pid >> 6;
  int h  = t & (NH - 1);
  int t2 = t >> 4;
  int s  = t2 & (SEQ - 1);
  int b  = t2 >> 11;
  if (b >= NB) return;
  float c  = fc[s * HALF + i];
  float sn = fs[s * HALF + i];
  size_t base = (size_t)(b * SEQ + s) * DIM + h * HD + 2 * i;   // even -> 4B aligned
  unsigned qp = *(unsigned*)(Q + base);
  float q0 = bf2f((unsigned short)qp), q1 = bf2f((unsigned short)(qp >> 16));
  unsigned kp = *(unsigned*)(Kt + base);
  float k0 = bf2f((unsigned short)kp), k1 = bf2f((unsigned short)(kp >> 16));
  float qo0 = q0 * c - q1 * sn, qo1 = q0 * sn + q1 * c;
  float ko0 = k0 * c - k1 * sn, ko1 = k0 * sn + k1 * c;
  *(unsigned*)(Q + base)  = (unsigned)f2bf(qo0) | ((unsigned)f2bf(qo1) << 16);
  *(unsigned*)(Kt + base) = (unsigned)f2bf(ko0) | ((unsigned)f2bf(ko1) << 16);
}

// ---------------------------------------------------------------------------
// Causal flash attention, all-bf16 I/O.  Block = 128 threads (4 waves),
// 64 q-rows per block, K/V streamed in 32-wide tiles up to the diagonal.
// ---------------------------------------------------------------------------
#define QROWS 64
#define KTILE 32
#define QLD   136   // 128 + 8 pad halves (272B = 17*16B)
#define VLD   40    // 32 + 8 pad halves
#define PLD   40

__global__ void __launch_bounds__(128)
flash_attn(const unsigned short* __restrict__ Q, const unsigned short* __restrict__ Kg,
           const unsigned short* __restrict__ V, unsigned short* __restrict__ O) {
  __shared__ alignas(16) unsigned short Qs[QROWS * QLD];
  __shared__ alignas(16) unsigned short Ks[KTILE * QLD];
  __shared__ alignas(16) unsigned short Vt[HD * VLD];
  __shared__ alignas(16) unsigned short Ps[4 * 16 * PLD];

  const int tid  = threadIdx.x;
  const int lane = tid & 31;
  const int w    = tid >> 5;
  const int lr   = lane & 15;
  const int lh   = lane >> 4;
  const int qt = blockIdx.x, h = blockIdx.y, b = blockIdx.z;
  const int qbase = qt * QROWS;
  const size_t headoff = (size_t)h * HD;
  const float sm_scale = 0.08838834764831845f;  // 1/sqrt(128)

  // Stage 64x128 Q tile via async copies (pure bf16 copy now).
#pragma unroll
  for (int t = 0; t < 8; ++t) {
    int idx = t * 128 + tid;            // 0..1023 (16B chunks)
    int r = idx >> 4, c8 = idx & 15;
    async_cp16(lds_off(Qs + r * QLD + c8 * 8),
               Q + (size_t)(b * SEQ + qbase + r) * DIM + headoff + c8 * 8);
  }
  wait_async0();
  __syncthreads();

  v16bf qa[4];
#pragma unroll
  for (int kk = 0; kk < 4; ++kk) {
    const unsigned short* p = Qs + (w * 16 + lr) * QLD + kk * 32;
    qa[kk] = load_frag(p, lh * 8, 16 + lh * 8);
  }

  v8f zero = {};
  v8f outacc[8];
#pragma unroll
  for (int nj = 0; nj < 8; ++nj) outacc[nj] = zero;
  float mrow[8], lrow[8];
#pragma unroll
  for (int r = 0; r < 8; ++r) { mrow[r] = -3.0e38f; lrow[r] = 0.0f; }

  const int ktmax = 2 * qt + 1;
  for (int kt = 0; kt <= ktmax; ++kt) {
    const int kbase = kt * KTILE;
    __syncthreads();
    // K tile: async copy 32x128 halves.
#pragma unroll
    for (int t = 0; t < 4; ++t) {
      int idx = t * 128 + tid;          // 0..511
      int r = idx >> 4, c8 = idx & 15;
      async_cp16(lds_off(Ks + r * QLD + c8 * 8),
                 Kg + (size_t)(b * SEQ + kbase + r) * DIM + headoff + c8 * 8);
    }
    // V tile: transpose while staging (8 halves per load, scattered b16 stores).
#pragma unroll
    for (int t = 0; t < 4; ++t) {
      int idx = t * 128 + tid;          // 0..511
      int kk = idx >> 4, d8 = idx & 15;
      v4u vv = *(const v4u*)(V + (size_t)(b * SEQ + kbase + kk) * DIM + headoff + d8 * 8);
#pragma unroll
      for (int j = 0; j < 4; ++j) {
        Vt[(d8 * 8 + 2 * j) * VLD + kk]     = (unsigned short)vv[j];
        Vt[(d8 * 8 + 2 * j + 1) * VLD + kk] = (unsigned short)(vv[j] >> 16);
      }
    }
    wait_async0();
    __syncthreads();

    // Scores S[16x32] = Qw[16x128] * K[32x128]^T
    v8f s0 = zero, s1 = zero;
#pragma unroll
    for (int kk = 0; kk < 4; ++kk) {
      const unsigned short* p0 = Ks + lr * QLD + kk * 32;
      v16bf b0 = load_frag(p0, lh * 8, 16 + lh * 8);
      s0 = wmma_bf16(qa[kk], b0, s0);
      const unsigned short* p1 = Ks + (16 + lr) * QLD + kk * 32;
      v16bf b1 = load_frag(p1, lh * 8, 16 + lh * 8);
      s1 = wmma_bf16(qa[kk], b1, s1);
    }

    const bool needmask = (kbase + KTILE - 1) > (qbase + w * 16);
    unsigned short* pw = Ps + w * 16 * PLD;
#pragma unroll
    for (int r = 0; r < 8; ++r) {
      int rowl = r + 8 * lh;
      int rowg = qbase + w * 16 + rowl;
      float x0 = s0[r] * sm_scale;
      float x1 = s1[r] * sm_scale;
      if (needmask) {
        if (kbase + lr > rowg)      x0 = -1.0e9f;
        if (kbase + 16 + lr > rowg) x1 = -1.0e9f;
      }
      float mx = fmaxf(x0, x1);
      mx = fmaxf(mx, __shfl_xor(mx, 1));
      mx = fmaxf(mx, __shfl_xor(mx, 2));
      mx = fmaxf(mx, __shfl_xor(mx, 4));
      mx = fmaxf(mx, __shfl_xor(mx, 8));
      float mnew = fmaxf(mrow[r], mx);
      float p0 = __expf(x0 - mnew);
      float p1 = __expf(x1 - mnew);
      float sum = p0 + p1;
      sum += __shfl_xor(sum, 1);
      sum += __shfl_xor(sum, 2);
      sum += __shfl_xor(sum, 4);
      sum += __shfl_xor(sum, 8);
      float corr = __expf(mrow[r] - mnew);
      lrow[r] = lrow[r] * corr + sum;
      mrow[r] = mnew;
#pragma unroll
      for (int nj = 0; nj < 8; ++nj) outacc[nj][r] *= corr;
      pw[rowl * PLD + lr]      = f2bf(p0);
      pw[rowl * PLD + 16 + lr] = f2bf(p1);
    }

    // out += P[16x32] * V[32x128]
    {
      const unsigned short* pp = pw + lr * PLD;
      v16bf pa = load_frag(pp, lh * 8, 16 + lh * 8);
#pragma unroll
      for (int nj = 0; nj < 8; ++nj) {
        const unsigned short* vp = Vt + (nj * 16 + lr) * VLD;
        v16bf vb = load_frag(vp, lh * 8, 16 + lh * 8);
        outacc[nj] = wmma_bf16(pa, vb, outacc[nj]);
      }
    }
  }

#pragma unroll
  for (int nj = 0; nj < 8; ++nj) {
    int d = nj * 16 + lr;
#pragma unroll
    for (int r = 0; r < 8; ++r) {
      int s = qbase + w * 16 + r + 8 * lh;
      O[(size_t)(b * SEQ + s) * DIM + headoff + d] = f2bf(outacc[nj][r] / lrow[r]);
    }
  }
}

// ---------------------------------------------------------------------------
extern "C" void kernel_launch(void* const* d_in, const int* in_sizes, int n_in,
                              void* d_out, int out_size, void* d_ws, size_t ws_size,
                              hipStream_t stream) {
  (void)in_sizes; (void)n_in; (void)out_size; (void)ws_size;
  const float* x  = (const float*)d_in[0];
  const float* fc = (const float*)d_in[1];
  const float* fs = (const float*)d_in[2];
  const float* wq = (const float*)d_in[3];
  const float* wk = (const float*)d_in[4];
  const float* wv = (const float*)d_in[5];
  const float* wo = (const float*)d_in[6];
  float* out = (float*)d_out;

  const size_t BSD = (size_t)NB * SEQ * DIM;   // 8,388,608
  const size_t DD  = (size_t)DIM * DIM;        // 4,194,304
  unsigned short* xb  = (unsigned short*)d_ws; // bf16 workspace (~117 MB total)
  unsigned short* wqb = xb  + BSD;
  unsigned short* wkb = wqb + DD;
  unsigned short* wvb = wkb + DD;
  unsigned short* wob = wvb + DD;
  unsigned short* Qb  = wob + DD;
  unsigned short* Kb  = Qb  + BSD;
  unsigned short* Vb  = Kb  + BSD;
  unsigned short* Ab  = Vb  + BSD;

  // 1) one-shot fp32 -> bf16 conversion of activations + weights
  convert_bf16<<<(int)(BSD / 8 / 256), 256, 0, stream>>>(x,  xb,  (int)(BSD / 8));
  convert_bf16<<<(int)(DD  / 8 / 256), 256, 0, stream>>>(wq, wqb, (int)(DD / 8));
  convert_bf16<<<(int)(DD  / 8 / 256), 256, 0, stream>>>(wk, wkb, (int)(DD / 8));
  convert_bf16<<<(int)(DD  / 8 / 256), 256, 0, stream>>>(wv, wvb, (int)(DD / 8));
  convert_bf16<<<(int)(DD  / 8 / 256), 256, 0, stream>>>(wo, wob, (int)(DD / 8));

  // 2) QKV projections (bf16 out)
  dim3 gg(DIM / GT_N, BS / GT_M);
  gemm_bf16<true><<<gg, 256, 0, stream>>>(xb, wqb, Qb, BS, DIM, DIM);
  gemm_bf16<true><<<gg, 256, 0, stream>>>(xb, wkb, Kb, BS, DIM, DIM);
  gemm_bf16<true><<<gg, 256, 0, stream>>>(xb, wvb, Vb, BS, DIM, DIM);

  // 3) RoPE in place on bf16 Q/K
  int pairs = NB * SEQ * NH * (HD / 2);
  rope_qk<<<pairs / 256, 256, 0, stream>>>(Qb, Kb, fc, fs);

  // 4) causal flash attention (bf16 in/out)
  flash_attn<<<dim3(SEQ / QROWS, NH, NB), 128, 0, stream>>>(Qb, Kb, Vb, Ab);

  // 5) output projection (fp32 out)
  gemm_bf16<false><<<gg, 256, 0, stream>>>(Ab, wob, out, BS, DIM, DIM);
}
